// GraphSAGE_23381801959787
// MI455X (gfx1250) — compile-verified, matrix operand order checked by reading, not run on hardware
//
#include <hip/hip_runtime.h>
#include <hip/hip_bf16.h>

#define N0 100000
#define N1 25000
#define N2 6250
#define N3 1600
#define DEG 16
#define D_IN 512
#define D_H 256

typedef __attribute__((ext_vector_type(16))) __bf16 v16bf;
typedef __attribute__((ext_vector_type(8)))  __bf16 v8bf;
typedef __attribute__((ext_vector_type(8)))  float  v8f;

static __device__ __forceinline__ __bf16 f2bf(float f) {
    union { float f; unsigned u; } v; v.f = f;
    unsigned r = v.u + 0x7FFFu + ((v.u >> 16) & 1u);   // round-to-nearest-even
    unsigned short s = (unsigned short)(r >> 16);
    union { unsigned short s; __bf16 b; } o; o.s = s;
    return o.b;
}

// ---------------------------------------------------------------- utility ---
__global__ void zero_f32(float* __restrict__ p, long n) {
    long i = (long)blockIdx.x * blockDim.x + threadIdx.x;
    if (i < n) p[i] = 0.0f;
}

__global__ void scatter_deg(const int* __restrict__ dst, float* __restrict__ deg, int E) {
    int i = blockIdx.x * blockDim.x + threadIdx.x;
    if (i < E) atomicAdd(&deg[dst[i]], 1.0f);
}

// one thread handles 4 consecutive features of one edge
__global__ void scatter_feat(const float* __restrict__ h, const int* __restrict__ src,
                             const int* __restrict__ dst, float* __restrict__ agg,
                             int E, int D) {
    long idx = (long)blockIdx.x * blockDim.x + threadIdx.x;
    int chunks = D >> 2;
    long e = idx / chunks;
    if (e >= E) return;
    int c = (int)(idx - e * chunks) << 2;
    int s = src[e], d = dst[e];
    const float4 v = *reinterpret_cast<const float4*>(h + (size_t)s * D + c);
    float* a = agg + (size_t)d * D + c;
    atomicAdd(a + 0, v.x);
    atomicAdd(a + 1, v.y);
    atomicAdd(a + 2, v.z);
    atomicAdd(a + 3, v.w);
}

__global__ void recip_deg(const float* __restrict__ deg, float* __restrict__ rdeg, int n) {
    int i = blockIdx.x * blockDim.x + threadIdx.x;
    if (i < n) rdeg[i] = 1.0f / fmaxf(deg[i], 1.0f);
}

// Build WT[n][k] = bf16( k < D ? W1[k,n] : W2[k-D,n] ), n in [0,256), k in [0,Ktot)
__global__ void wprep(const float* __restrict__ W1, const float* __restrict__ W2,
                      __bf16* __restrict__ WT, int D, int Ktot) {
    long idx = (long)blockIdx.x * blockDim.x + threadIdx.x;
    long total = (long)256 * Ktot;
    if (idx >= total) return;
    int n = (int)(idx / Ktot);
    int k = (int)(idx - (long)n * Ktot);
    float v = (k < D) ? W1[(size_t)k * 256 + n] : W2[(size_t)(k - D) * 256 + n];
    WT[idx] = f2bf(v);
}

// ------------------------------------------------------------- fused GEMM ---
// C[M x 256] = act( [A1 | A2*rdeg] @ W + bias ),  W given pre-transposed bf16:
// WT[n][k], n in [0,256), k in [0,Ktot). Ktot = D (A2==null) or 2*D.
// Block: 256 threads (8 wave32), 64x64 output tile, K-step 32, LDS ping-pong.
__global__ __launch_bounds__(256)
void sage_gemm(const float* __restrict__ A1, const float* __restrict__ A2,
               const float* __restrict__ rdeg,
               const __bf16* __restrict__ WT,
               const float* __restrict__ bias, float* __restrict__ C,
               int M, int D, int Ktot, int relu) {
    __shared__ __bf16 As[2][64][32];    // [buf][m][k]
    __shared__ __bf16 BsT[2][64][32];   // [buf][n][k]

    const int t    = threadIdx.x;
    const int wid  = t >> 5;         // 0..7
    const int lane = t & 31;
    const int wm   = wid >> 1;       // 0..3  -> M sub-tile (16 rows)
    const int wn   = wid & 1;        // 0..1  -> N sub-tile (32 cols)
    const int half = lane >> 4;      // 0/1
    const int l15  = lane & 15;

    const int blockM = blockIdx.x * 64;
    const int blockN = blockIdx.y * 64;

    // A loader: 8 consecutive k of one row per thread
    const int arow = t >> 2;                 // 0..63
    const int acol = (t & 3) * 8;            // 0,8,16,24
    const int grow = blockM + arow;
    // B loader: 8 consecutive k of one WT row per thread
    const int nrow = t >> 2;                 // 0..63
    const int kc   = (t & 3) * 8;            // 0,8,16,24

    const float arscale = (A2 && grow < M) ? rdeg[grow] : 1.0f;

    auto stageA = [&](int buf, int k0) {
        float vals[8];
        int k = k0 + acol;
        if (grow < M) {
            const float* sp;
            float scale = 1.0f;
            if (k < D) {
                sp = A1 + (size_t)grow * D + k;
            } else {
                sp = A2 + (size_t)grow * D + (k - D);
                scale = arscale;
            }
            const float4 v0 = *reinterpret_cast<const float4*>(sp);
            const float4 v1 = *reinterpret_cast<const float4*>(sp + 4);
            vals[0] = v0.x * scale; vals[1] = v0.y * scale;
            vals[2] = v0.z * scale; vals[3] = v0.w * scale;
            vals[4] = v1.x * scale; vals[5] = v1.y * scale;
            vals[6] = v1.z * scale; vals[7] = v1.w * scale;
        } else {
#pragma unroll
            for (int j = 0; j < 8; ++j) vals[j] = 0.0f;
        }
        v8bf pk;
#pragma unroll
        for (int j = 0; j < 8; ++j) pk[j] = f2bf(vals[j]);
        *reinterpret_cast<v8bf*>(&As[buf][arow][acol]) = pk;
    };

    auto stageB = [&](int buf, int k0) {
        const v8bf w = *reinterpret_cast<const v8bf*>(
            WT + (size_t)(blockN + nrow) * Ktot + k0 + kc);
        *reinterpret_cast<v8bf*>(&BsT[buf][nrow][kc]) = w;
    };

    v8f acc[2] = { {}, {} };
    const int niter = Ktot >> 5;

    stageA(0, 0);
    stageB(0, 0);
    __syncthreads();

    const int m   = wm * 16 + l15;
    const int kb0 = half * 8;
    const int kb1 = 16 + half * 8;

    for (int it = 0; it < niter; ++it) {
        const int cur = it & 1;
        if (it + 1 < niter) {           // prefetch next tile into other buffer
            stageA(cur ^ 1, (it + 1) << 5);
            stageB(cur ^ 1, (it + 1) << 5);
        }

        // A frag: lane(0-15): K 0-7 & 16-23 ; lane(16-31): K 8-15 & 24-31
        v8bf alo = *reinterpret_cast<const v8bf*>(&As[cur][m][kb0]);
        v8bf ahi = *reinterpret_cast<const v8bf*>(&As[cur][m][kb1]);
        v16bf a;
#pragma unroll
        for (int j = 0; j < 8; ++j) { a[j] = alo[j]; a[j + 8] = ahi[j]; }

#pragma unroll
        for (int s = 0; s < 2; ++s) {
            const int n = wn * 32 + s * 16 + l15;
            v8bf blo = *reinterpret_cast<const v8bf*>(&BsT[cur][n][kb0]);
            v8bf bhi = *reinterpret_cast<const v8bf*>(&BsT[cur][n][kb1]);
            v16bf b;
#pragma unroll
            for (int j = 0; j < 8; ++j) { b[j] = blo[j]; b[j + 8] = bhi[j]; }
            acc[s] = __builtin_amdgcn_wmma_f32_16x16x32_bf16(
                false, a, false, b, (short)0, acc[s], false, false);
        }
        __syncthreads();
    }

    // ---- epilogue: bias (+ReLU), f32 store ----
#pragma unroll
    for (int s = 0; s < 2; ++s) {
        const int col = blockN + wn * 32 + s * 16 + l15;
        const float bv = bias[col];
#pragma unroll
        for (int r = 0; r < 8; ++r) {
            const int row = blockM + wm * 16 + half * 8 + r;
            if (row < M) {
                float v = acc[s][r] + bv;
                if (relu) v = fmaxf(v, 0.0f);
                C[(size_t)row * 256 + col] = v;
            }
        }
    }
}

// ---------------------------------------------------------------- driver ----
static inline int cdiv(long a, long b) { return (int)((a + b - 1) / b); }

extern "C" void kernel_launch(void* const* d_in, const int* in_sizes, int n_in,
                              void* d_out, int out_size, void* d_ws, size_t ws_size,
                              hipStream_t stream) {
    const float* x    = (const float*)d_in[0];
    const int*   src0 = (const int*)d_in[1];
    const int*   dst0 = (const int*)d_in[2];
    const int*   src1 = (const int*)d_in[3];
    const int*   dst1 = (const int*)d_in[4];
    const int*   src2 = (const int*)d_in[5];
    const int*   dst2 = (const int*)d_in[6];
    const float* Ws0  = (const float*)d_in[7];
    const float* Wn0  = (const float*)d_in[8];
    const float* b0   = (const float*)d_in[9];
    const float* Ws1  = (const float*)d_in[10];
    const float* Wn1  = (const float*)d_in[11];
    const float* b1   = (const float*)d_in[12];
    const float* Ws2  = (const float*)d_in[13];
    const float* Wn2  = (const float*)d_in[14];
    const float* b2   = (const float*)d_in[15];
    const float* Wfc  = (const float*)d_in[16];
    const float* bfc  = (const float*)d_in[17];
    float* out = (float*)d_out;

    // workspace carve (floats)
    float*  ws   = (float*)d_ws;
    float*  h1   = ws;                         // N1 x 256
    float*  h2   = h1 + (size_t)N1 * D_H;      // N2 x 256
    float*  h3   = h2 + (size_t)N2 * D_H;      // N3 x 256
    float*  agg  = h3 + (size_t)N3 * D_H;      // up to N1 x 512
    float*  deg  = agg + (size_t)N1 * D_IN;    // N1
    float*  rdeg = deg + N1;                   // N1
    __bf16* WT   = (__bf16*)(rdeg + N1);       // 256 x 1024 bf16 max

    const int B = 256;

    // ---------------- layer 0: x (N0 x 512) -> h1 (N1 x 256), relu ----------
    {
        const int D = D_IN, Ktot = 2 * D_IN;
        long na = (long)N1 * D;
        zero_f32<<<cdiv(na, B), B, 0, stream>>>(agg, na);
        zero_f32<<<cdiv(N1, B), B, 0, stream>>>(deg, N1);
        int E = N1 * DEG;
        scatter_deg<<<cdiv(E, B), B, 0, stream>>>(dst0, deg, E);
        long nt = (long)E * (D >> 2);
        scatter_feat<<<cdiv(nt, B), B, 0, stream>>>(x, src0, dst0, agg, E, D);
        recip_deg<<<cdiv(N1, B), B, 0, stream>>>(deg, rdeg, N1);
        wprep<<<cdiv((long)256 * Ktot, B), B, 0, stream>>>(Ws0, Wn0, WT, D, Ktot);
        dim3 g(cdiv(N1, 64), 4);
        sage_gemm<<<g, B, 0, stream>>>(x, agg, rdeg, WT, b0, h1, N1, D, Ktot, 1);
    }
    // ---------------- layer 1: h1 -> h2 (N2 x 256), relu --------------------
    {
        const int D = D_H, Ktot = 2 * D_H;
        long na = (long)N2 * D;
        zero_f32<<<cdiv(na, B), B, 0, stream>>>(agg, na);
        zero_f32<<<cdiv(N2, B), B, 0, stream>>>(deg, N2);
        int E = N2 * DEG;
        scatter_deg<<<cdiv(E, B), B, 0, stream>>>(dst1, deg, E);
        long nt = (long)E * (D >> 2);
        scatter_feat<<<cdiv(nt, B), B, 0, stream>>>(h1, src1, dst1, agg, E, D);
        recip_deg<<<cdiv(N2, B), B, 0, stream>>>(deg, rdeg, N2);
        wprep<<<cdiv((long)256 * Ktot, B), B, 0, stream>>>(Ws1, Wn1, WT, D, Ktot);
        dim3 g(cdiv(N2, 64), 4);
        sage_gemm<<<g, B, 0, stream>>>(h1, agg, rdeg, WT, b1, h2, N2, D, Ktot, 1);
    }
    // ---------------- layer 2: h2 -> h3 (N3 x 256), no relu -----------------
    {
        const int D = D_H, Ktot = 2 * D_H;
        long na = (long)N3 * D;
        zero_f32<<<cdiv(na, B), B, 0, stream>>>(agg, na);
        zero_f32<<<cdiv(N3, B), B, 0, stream>>>(deg, N3);
        int E = N3 * DEG;
        scatter_deg<<<cdiv(E, B), B, 0, stream>>>(dst2, deg, E);
        long nt = (long)E * (D >> 2);
        scatter_feat<<<cdiv(nt, B), B, 0, stream>>>(h2, src2, dst2, agg, E, D);
        recip_deg<<<cdiv(N3, B), B, 0, stream>>>(deg, rdeg, N3);
        wprep<<<cdiv((long)256 * Ktot, B), B, 0, stream>>>(Ws2, Wn2, WT, D, Ktot);
        dim3 g(cdiv(N3, 64), 4);
        sage_gemm<<<g, B, 0, stream>>>(h2, agg, rdeg, WT, b2, h3, N3, D, Ktot, 0);
    }
    // ---------------- FC: out = h3 @ W_fc + b_fc ----------------------------
    {
        const int D = D_H, Ktot = D_H;
        wprep<<<cdiv((long)256 * Ktot, B), B, 0, stream>>>(Wfc, nullptr, WT, D, Ktot);
        dim3 g(cdiv(N3, 64), 4);
        sage_gemm<<<g, B, 0, stream>>>(h3, nullptr, nullptr, WT, bfc, out, N3, D, Ktot, 0);
    }
}
